// VectorQuantizer_2946347565094
// MI455X (gfx1250) — matrix operand order repaired and checked by reading, not compiled, not run on hardware
//
#include <hip/hip_runtime.h>
#include <hip/hip_bf16.h>

// CDNA5 / gfx1250 vector-quantizer:
//   dist(row, code) = ||e||^2 - 2 x.e   (||x||^2 dropped: constant per row)
// A = bf16(-2x) [16x32] in registers (loop-invariant), B = bf16 codebook
// staged once per block into LDS via global_load_async_to_lds_b128,
// C = broadcast(||e||^2).  Two independent WMMA accumulator chains per
// iteration keep the XDL pipe busy while min/argmin VALU co-executes.

typedef __attribute__((ext_vector_type(16))) __bf16 v16bf;
typedef __attribute__((ext_vector_type(8)))  __bf16 v8bf;
typedef __attribute__((ext_vector_type(8)))  float  v8f;
typedef __attribute__((address_space(3))) unsigned short lds_us_t;

#define K_CODES  1024
#define DIM      64
#define HW       4096        // H*W per image
#define TILE_M   128         // pixels per block
#define N_ELEM   4194304.0f  // N * D = 65536 * 64
#define CB_PITCH 72          // LDS row pitch (ushorts): 144 B -> bank-conflict-free
#define CB_BYTES (K_CODES * CB_PITCH * 2)
#define SMEM_BYTES (CB_BYTES + TILE_M * 4 + 8 * 4)

union V16U { v16bf v; v8bf h[2]; unsigned short s[16]; };

__device__ __forceinline__ unsigned short f32_to_bf16_rne(float f) {
  unsigned int u = __float_as_uint(f);
  u += 0x7FFFu + ((u >> 16) & 1u);   // round-to-nearest-even
  return (unsigned short)(u >> 16);
}

// ---------------------------------------------------------------------------
// Kernel 0: codebook -> bf16 row-major table + per-code squared norms.
// ---------------------------------------------------------------------------
__global__ void vq_prep(const float* __restrict__ emb,
                        unsigned short* __restrict__ ebf,
                        float* __restrict__ enorm) {
  int k = blockIdx.x * blockDim.x + threadIdx.x;
  if (k >= K_CODES) return;
  float acc = 0.0f;
  #pragma unroll 8
  for (int d = 0; d < DIM; ++d) {
    float w = emb[k * DIM + d];
    acc += w * w;
    ebf[k * DIM + d] = f32_to_bf16_rne(w);
  }
  enorm[k] = acc;
}

// ---------------------------------------------------------------------------
// Kernel 1: 128-pixel tile: async-stage codebook to LDS, WMMA distances,
// argmin, gather, NT output store, deterministic loss partial.
// ---------------------------------------------------------------------------
__global__ void __launch_bounds__(256)
vq_main(const float* __restrict__ xin,
        const float* __restrict__ emb,
        const unsigned short* __restrict__ ebf,
        const float* __restrict__ enorm,
        float* __restrict__ outq,        // d_out + 1 (the [B,D,H,W] output)
        float* __restrict__ partials) {
  extern __shared__ unsigned char smem[];
  unsigned short* lds_cb  = (unsigned short*)smem;           // 1024 x 72 us
  int*            lds_idx = (int*)(smem + CB_BYTES);         // 128 ints
  float*          lds_part= (float*)(smem + CB_BYTES + TILE_M * 4);

  const int tid  = threadIdx.x;
  const int lane = tid & 31;
  const int wave = tid >> 5;     // 0..7
  const int hl   = lane >> 4;    // half of wave (0/1)
  const int lm   = lane & 15;    // column class / row-in-half

  const int p_tile = blockIdx.x * TILE_M;
  const int b      = p_tile >> 12;          // image index (4096 px per image)
  const int p_in_b = p_tile & (HW - 1);

  // ---- async-stage the bf16 codebook (128 KB) into LDS, padded rows.
  // Each of 1024 rows: 128 B global -> 144 B-pitch LDS, 8 x b128 per row.
  for (int r = tid; r < K_CODES; r += 256) {
    unsigned loff = (unsigned)(size_t)(lds_us_t*)&lds_cb[r * CB_PITCH];
    unsigned long long g = (unsigned long long)(const void*)(ebf + r * DIM);
    #pragma unroll
    for (int j = 0; j < 8; ++j) {
      asm volatile("global_load_async_to_lds_b128 %0, %1, off"
                   :: "v"(loff + 16u * j), "v"(g + 16ull * j) : "memory");
    }
  }

  // ---- A fragments (overlap the async DMA): this wave's 16 pixel rows,
  // f32 -> bf16(-2x), contiguous k = 16h + s order (same as B):
// lanes 0-15: K = 0..15 (s[0..15]); lanes 16-31: K = 16..31.
  const int m = wave * 16 + lm;                         // row within tile
  const size_t xbase = (size_t)b * DIM * HW + (size_t)(p_in_b + m);
  const int koff = hl * 16;
  V16U A[2];
  #pragma unroll
  for (int kk = 0; kk < 2; ++kk) {
    #pragma unroll
    for (int j = 0; j < 8; ++j) {
      int d0 = kk * 32 + koff + j;
      int d1 = d0 + 8;
      float x0 = xin[xbase + (size_t)d0 * HW];
      float x1 = xin[xbase + (size_t)d1 * HW];
      A[kk].s[j]     = f32_to_bf16_rne(-2.0f * x0);
      A[kk].s[8 + j] = f32_to_bf16_rne(-2.0f * x1);
    }
  }

  asm volatile("s_wait_asynccnt 0x0" ::: "memory");
  __syncthreads();

  float bestv[8];
  int   besti[8];
  #pragma unroll
  for (int j = 0; j < 8; ++j) { bestv[j] = 3.4e38f; besti[j] = 0; }

  // ---- stream the 1024 codes, 32 per iteration: two independent WMMA
  // accumulator chains so consecutive wmma issues have no RAW dependence.
  const int dB = hl * 16;   // B K-offset: lanes 0-15 -> K 0..15, 16-31 -> 16..31
  for (int cb = 0; cb < K_CODES; cb += 32) {
    const unsigned short* row0 = &lds_cb[(cb + lm) * CB_PITCH];        // cols cb+lm
    const unsigned short* row1 = &lds_cb[(cb + 16 + lm) * CB_PITCH];   // cols cb+16+lm
    const float env0 = enorm[cb + lm];
    const float env1 = enorm[cb + 16 + lm];

    V16U b0a, b1a, b0b, b1b;   // (kk, chunk)
    b0a.h[0] = *(const v8bf*)(row0 + dB);      b0a.h[1] = *(const v8bf*)(row0 + dB + 8);
    b1a.h[0] = *(const v8bf*)(row0 + 32 + dB); b1a.h[1] = *(const v8bf*)(row0 + 32 + dB + 8);
    b0b.h[0] = *(const v8bf*)(row1 + dB);      b0b.h[1] = *(const v8bf*)(row1 + dB + 8);
    b1b.h[0] = *(const v8bf*)(row1 + 32 + dB); b1b.h[1] = *(const v8bf*)(row1 + 32 + dB + 8);

    v8f c0, c1;
    #pragma unroll
    for (int j = 0; j < 8; ++j) { c0[j] = env0; c1[j] = env1; }

    c0 = __builtin_amdgcn_wmma_f32_16x16x32_bf16(false, A[0].v, false, b0a.v,
                                                 (short)0, c0, false, false);
    c1 = __builtin_amdgcn_wmma_f32_16x16x32_bf16(false, A[0].v, false, b0b.v,
                                                 (short)0, c1, false, false);
    c0 = __builtin_amdgcn_wmma_f32_16x16x32_bf16(false, A[1].v, false, b1a.v,
                                                 (short)0, c0, false, false);
    c1 = __builtin_amdgcn_wmma_f32_16x16x32_bf16(false, A[1].v, false, b1b.v,
                                                 (short)0, c1, false, false);

    const int col0 = cb + lm;
    const int col1 = cb + 16 + lm;
    #pragma unroll
    for (int j = 0; j < 8; ++j) {
      if (c0[j] < bestv[j]) { bestv[j] = c0[j]; besti[j] = col0; }
    }
    #pragma unroll
    for (int j = 0; j < 8; ++j) {
      if (c1[j] < bestv[j]) { bestv[j] = c1[j]; besti[j] = col1; }
    }
  }

  // ---- argmin across the 16 column-class lanes (masks < 16 stay in-half),
  // first-index tie-break to match jnp.argmin.
  #pragma unroll
  for (int j = 0; j < 8; ++j) {
    #pragma unroll
    for (int msk = 8; msk >= 1; msk >>= 1) {
      float ov = __shfl_xor(bestv[j], msk, 32);
      int   oi = __shfl_xor(besti[j], msk, 32);
      if (ov < bestv[j] || (ov == bestv[j] && oi < besti[j])) {
        bestv[j] = ov; besti[j] = oi;
      }
    }
  }
  if (lm == 0) {
    // lane 0 -> rows wave*16 + 0..7 ; lane 16 -> rows wave*16 + 8..15
    #pragma unroll
    for (int j = 0; j < 8; ++j)
      lds_idx[wave * 16 + hl * 8 + j] = besti[j];
  }
  __syncthreads();

  // ---- gather codebook rows, write [B,D,H,W] output (streamed, NT stores),
  // accumulate squared error for the losses.
  float acc = 0.0f;
  #pragma unroll 4
  for (int it = 0; it < 32; ++it) {
    int e = tid + 256 * it;          // 0..8191 over (d, p)
    int d = e >> 7;                  // 0..63
    int p = e & (TILE_M - 1);        // consecutive lanes -> consecutive p
    int code = lds_idx[p];
    float q  = emb[code * DIM + d];  // gather; 256 KB table is cache-resident
    size_t off = (size_t)b * DIM * HW + (size_t)d * HW + (size_t)(p_in_b + p);
    float xv = xin[off];
    __builtin_nontemporal_store(q, &outq[off]);
    float diff = q - xv;
    acc += diff * diff;
  }

  // deterministic block reduction (no float atomics -> replay-stable)
  #pragma unroll
  for (int msk = 16; msk >= 1; msk >>= 1)
    acc += __shfl_xor(acc, msk, 32);
  if (lane == 0) lds_part[wave] = acc;
  __syncthreads();
  if (tid == 0) {
    float s = 0.0f;
    #pragma unroll
    for (int w = 0; w < 8; ++w) s += lds_part[w];
    partials[blockIdx.x] = s;
  }
}

// ---------------------------------------------------------------------------
// Kernel 2: fold 512 partials -> loss scalars (fixed order, deterministic).
// ---------------------------------------------------------------------------
__global__ void __launch_bounds__(512)
vq_finalize(const float* __restrict__ partials, float* __restrict__ out) {
  __shared__ float sh[16];
  int tid = threadIdx.x;            // 512 threads = 16 waves
  float v = partials[tid];
  #pragma unroll
  for (int msk = 16; msk >= 1; msk >>= 1) v += __shfl_xor(v, msk, 32);
  if ((tid & 31) == 0) sh[tid >> 5] = v;
  __syncthreads();
  if (tid == 0) {
    float s = 0.0f;
    #pragma unroll
    for (int w = 0; w < 16; ++w) s += sh[w];
    float mse = s / N_ELEM;
    out[0] = 1.25f * mse;            // loss = q_latent + 0.25 * e_latent
    out[1 + 4194304] = mse;          // e_latent_loss
    out[2 + 4194304] = mse;          // q_latent_loss
  }
}

extern "C" void kernel_launch(void* const* d_in, const int* in_sizes, int n_in,
                              void* d_out, int out_size, void* d_ws, size_t ws_size,
                              hipStream_t stream) {
  const float* xin = (const float*)d_in[0];   // [16, 64, 64, 64] f32
  const float* emb = (const float*)d_in[1];   // [1024, 64] f32
  float* out = (float*)d_out;                 // [loss | 16*64*64*64 | e | q]

  // workspace: bf16 codebook (128 KB) | code norms (4 KB) | 512 partials (2 KB)
  unsigned short* ebf   = (unsigned short*)d_ws;
  float*          enorm = (float*)((char*)d_ws + 131072);
  float*          parts = (float*)((char*)d_ws + 131072 + 4096);

  vq_prep<<<4, 256, 0, stream>>>(emb, ebf, enorm);
  vq_main<<<512, 256, SMEM_BYTES, stream>>>(xin, emb, ebf, enorm, out + 1, parts);
  vq_finalize<<<1, 512, 0, stream>>>(parts, out);
}
